// MultiheadLatentAttention_69312182223049
// MI455X (gfx1250) — matrix operand order repaired and checked by reading, not compiled, not run on hardware
//
#include <hip/hip_runtime.h>
#include <math.h>

typedef __attribute__((ext_vector_type(8)))  float        v8f;
typedef __attribute__((ext_vector_type(16))) __bf16       v16bf;
typedef __attribute__((ext_vector_type(4)))  unsigned int u32x4;
typedef __attribute__((ext_vector_type(4)))  float        f32x4;

#define D_MODEL 2048
#define N_HEADS 16
#define HEAD_DIM 128
#define KV_DIM 512
#define Q_DIM 1536
#define ROPE_DIM 64
#define QK_DIM 192      /* HEAD_DIM + ROPE_DIM */
#define BB 2
#define SS 2048
#define BS (BB*SS)

__device__ __forceinline__ unsigned short f2bf(float f) {
  union { float f; unsigned int u; } v; v.f = f;
  unsigned int r = v.u + 0x7FFFu + ((v.u >> 16) & 1u);   // round-to-nearest-even
  return (unsigned short)(r >> 16);
}
__device__ __forceinline__ float bf2f(unsigned short h) {
  union { unsigned int u; float f; } v; v.u = ((unsigned int)h) << 16;
  return v.f;
}

union Frag { v16bf v; u32x4 u[2]; };

/* Async global -> LDS copy of 16 bytes (gfx1250 ASYNCcnt path).      */
/* LDS address = low 32 bits of the flat pointer (ISA: LDS aperture   */
/* maps an address to its low 32 bits).                                       */
__device__ __forceinline__ void async_copy_b128(const void* gptr, void* lptr) {
  unsigned lds_addr = (unsigned)(size_t)lptr;
  unsigned long long gaddr = (unsigned long long)(size_t)gptr;
  asm volatile("global_load_async_to_lds_b128 %0, %1, off"
               :: "v"(lds_addr), "v"(gaddr) : "memory");
}
__device__ __forceinline__ void wait_async_all() {
  asm volatile("s_wait_asynccnt 0" ::: "memory");
}

/* ------------------------------------------------------------------ */
/* fp32 -> bf16 conversion                                            */
/* ------------------------------------------------------------------ */
__global__ void cvt_f32_bf16(const float* __restrict__ x,
                             unsigned short* __restrict__ y, int n) {
  int i = blockIdx.x * 256 + threadIdx.x;
  if (i < n) y[i] = f2bf(x[i]);
}

/* ------------------------------------------------------------------ */
/* Generic bf16 GEMM:  C(MxN) = A(MxK,bf16) @ W(KxN,fp32) + bias      */
/* out_mode 0: bf16, column remap col=(n/seg)*stride + n%seg + off    */
/* out_mode 1: bf16 transposed store  C[n*ldc + m] (for V^T)          */
/* out_mode 2: fp32 row-major store                                   */
/* Block tile 128x128x32, 8 waves (4x2), wave tile 32x64 (8 WMMAs).   */
/* A tile staged via global_load_async_to_lds_b128; W tile converted  */
/* fp32->bf16 in-flight and stored transposed.                        */
/* ------------------------------------------------------------------ */
__global__ __launch_bounds__(256)
void gemm_bf16(const unsigned short* __restrict__ A, int lda,
               const float* __restrict__ W, int ldw,
               const float* __restrict__ bias,
               void* __restrict__ C, int ldc,
               int M, int N, int K,
               int col_seg, int col_stride, int col_off,
               int out_mode) {
  __shared__ __align__(16) unsigned short As[128 * 32];
  __shared__ __align__(16) unsigned short Wt[128 * 32];   // transposed: [n][k]

  const int tid    = threadIdx.x;
  const int wave   = tid >> 5, lane = tid & 31;
  const int lane16 = lane & 15, half = lane >> 4;
  const int wm = wave >> 1, wn = wave & 1;
  const int m0 = blockIdx.y * 128;
  const int n0 = blockIdx.x * 128;

  const v8f vzero = {0.f,0.f,0.f,0.f,0.f,0.f,0.f,0.f};
  v8f acc[2][4];
#pragma unroll
  for (int mi = 0; mi < 2; ++mi)
#pragma unroll
    for (int ni = 0; ni < 4; ++ni) acc[mi][ni] = vzero;

  for (int k0 = 0; k0 < K; k0 += 32) {
    // stage A (bf16, row-major 128x32) via async global->LDS DMA
#pragma unroll
    for (int i = 0; i < 2; ++i) {
      int idx = tid + i * 256;
      int row = idx >> 2, seg = idx & 3;
      async_copy_b128(A + (size_t)(m0 + row) * lda + k0 + seg * 8,
                      &As[row * 32 + seg * 8]);
    }
    // stage W: fp32 -> bf16, transposed into Wt[n][k]
#pragma unroll
    for (int i = 0; i < 4; ++i) {
      int idx = tid + i * 256;
      int kk = idx >> 5, nseg = idx & 31;
      int gn = n0 + nseg * 4;
      f32x4 w4 = {0.f, 0.f, 0.f, 0.f};
      if (gn < N) w4 = *(const f32x4*)(W + (size_t)(k0 + kk) * ldw + gn);
      Wt[(nseg * 4 + 0) * 32 + kk] = f2bf(w4[0]);
      Wt[(nseg * 4 + 1) * 32 + kk] = f2bf(w4[1]);
      Wt[(nseg * 4 + 2) * 32 + kk] = f2bf(w4[2]);
      Wt[(nseg * 4 + 3) * 32 + kk] = f2bf(w4[3]);
    }
    wait_async_all();
    __syncthreads();

    Frag af[2], bfr[4];
#pragma unroll
    for (int mi = 0; mi < 2; ++mi) {
      int row = wm * 32 + mi * 16 + lane16;
      af[mi].u[0] = *(const u32x4*)(&As[row * 32 + half * 8]);
      af[mi].u[1] = *(const u32x4*)(&As[row * 32 + 16 + half * 8]);
    }
#pragma unroll
    for (int ni = 0; ni < 4; ++ni) {
      int col = wn * 64 + ni * 16 + lane16;
      bfr[ni].u[0] = *(const u32x4*)(&Wt[col * 32 + half * 8]);
      bfr[ni].u[1] = *(const u32x4*)(&Wt[col * 32 + 16 + half * 8]);
    }
#pragma unroll
    for (int mi = 0; mi < 2; ++mi)
#pragma unroll
      for (int ni = 0; ni < 4; ++ni)
        acc[mi][ni] = __builtin_amdgcn_wmma_f32_16x16x32_bf16(
            false, af[mi].v, false, bfr[ni].v, (short)0, acc[mi][ni], false, false);
    __syncthreads();
  }

  // epilogue
#pragma unroll
  for (int mi = 0; mi < 2; ++mi) {
#pragma unroll
    for (int ni = 0; ni < 4; ++ni) {
      int n_raw = n0 + wn * 64 + ni * 16 + lane16;
      if (n_raw >= N) continue;
      float bv = bias ? bias[n_raw] : 0.f;
      v8f a = acc[mi][ni];
      int mbase = m0 + wm * 32 + mi * 16 + 8 * half;
      if (out_mode == 0) {
        int colr = (n_raw / col_seg) * col_stride + (n_raw % col_seg) + col_off;
        unsigned short* Cs = (unsigned short*)C;
#pragma unroll
        for (int j = 0; j < 8; ++j)
          Cs[(size_t)(mbase + j) * ldc + colr] = f2bf(a[j] + bv);
      } else if (out_mode == 1) {
        unsigned int w0 = (unsigned int)f2bf(a[0] + bv) | ((unsigned int)f2bf(a[1] + bv) << 16);
        unsigned int w1 = (unsigned int)f2bf(a[2] + bv) | ((unsigned int)f2bf(a[3] + bv) << 16);
        unsigned int w2 = (unsigned int)f2bf(a[4] + bv) | ((unsigned int)f2bf(a[5] + bv) << 16);
        unsigned int w3 = (unsigned int)f2bf(a[6] + bv) | ((unsigned int)f2bf(a[7] + bv) << 16);
        u32x4 pk = {w0, w1, w2, w3};
        *(u32x4*)((unsigned short*)C + (size_t)n_raw * ldc + mbase) = pk;
      } else {
        float* Cf = (float*)C;
#pragma unroll
        for (int j = 0; j < 8; ++j)
          Cf[(size_t)(mbase + j) * ldc + n_raw] = a[j] + bv;
      }
    }
  }
}

/* ------------------------------------------------------------------ */
/* Interleaved RoPE on qf rope slice (in place, per head)             */
/* ------------------------------------------------------------------ */
__global__ void rope_q(unsigned short* __restrict__ qf) {
  int idx = blockIdx.x * 256 + threadIdx.x;   // BS * 16 * 32
  int p   = idx & 31;
  int h   = (idx >> 5) & 15;
  int row = idx >> 9;
  if (row >= BS) return;
  int s = row & (SS - 1);
  float freq = __expf(-(float)(2 * p) * 0.07195583f);   // ln(10000)/128
  float ang = (float)s * freq;
  float sn, c; __sincosf(ang, &sn, &c);
  unsigned int* ptr = (unsigned int*)(qf + (size_t)row * (N_HEADS * QK_DIM)
                                      + h * QK_DIM + HEAD_DIM + 2 * p);
  unsigned int u = *ptr;
  float x1 = bf2f((unsigned short)(u & 0xFFFF));
  float x2 = bf2f((unsigned short)(u >> 16));
  float o1 = x1 * c - x2 * sn;
  float o2 = x1 * sn + x2 * c;
  *ptr = (unsigned int)f2bf(o1) | ((unsigned int)f2bf(o2) << 16);
}

/* RoPE on rope_k then broadcast to all 16 head slots of kf           */
__global__ void rope_k_bcast(const unsigned short* __restrict__ rk,
                             unsigned short* __restrict__ kf) {
  int idx = blockIdx.x * 256 + threadIdx.x;   // BS * 32
  int p   = idx & 31;
  int row = idx >> 5;
  if (row >= BS) return;
  int s = row & (SS - 1);
  float freq = __expf(-(float)(2 * p) * 0.07195583f);
  float ang = (float)s * freq;
  float sn, c; __sincosf(ang, &sn, &c);
  unsigned int u = *(const unsigned int*)(rk + (size_t)row * ROPE_DIM + 2 * p);
  float x1 = bf2f((unsigned short)(u & 0xFFFF));
  float x2 = bf2f((unsigned short)(u >> 16));
  float o1 = x1 * c - x2 * sn;
  float o2 = x1 * sn + x2 * c;
  unsigned int packed = (unsigned int)f2bf(o1) | ((unsigned int)f2bf(o2) << 16);
#pragma unroll
  for (int h = 0; h < N_HEADS; ++h)
    *(unsigned int*)(kf + (size_t)row * (N_HEADS * QK_DIM)
                     + h * QK_DIM + HEAD_DIM + 2 * p) = packed;
}

/* ------------------------------------------------------------------ */
/* Fused causal flash attention. Block = (b, h, 128 q rows), 8 waves, */
/* each wave owns 16 q rows. Q·K^T and P·V via bf16 WMMA; online      */
/* softmax with shfl reductions; P staged through per-wave LDS.       */
/* ------------------------------------------------------------------ */
__global__ __launch_bounds__(256)
void mla_attn(const unsigned short* __restrict__ qf,
              const unsigned short* __restrict__ kf,
              const unsigned short* __restrict__ vt,
              unsigned short* __restrict__ attn) {
  __shared__ __align__(16) unsigned short P[8 * 16 * 32];
  const int tid    = threadIdx.x;
  const int wave   = tid >> 5, lane = tid & 31;
  const int lane16 = lane & 15, half = lane >> 4;
  const int b = blockIdx.z, hIdx = blockIdx.y;
  const int qb = blockIdx.x * 128 + wave * 16;
  unsigned short* Pw = &P[wave * 16 * 32];

  // preload Q fragments (16 rows x 192 features = 6 K-chunks of 32)
  Frag qa[6];
  {
    size_t base = (size_t)(b * SS + qb + lane16) * (N_HEADS * QK_DIM) + hIdx * QK_DIM;
#pragma unroll
    for (int c = 0; c < 6; ++c) {
      qa[c].u[0] = *(const u32x4*)(qf + base + c * 32 + half * 8);
      qa[c].u[1] = *(const u32x4*)(qf + base + c * 32 + 16 + half * 8);
    }
  }

  const v8f vzero = {0.f,0.f,0.f,0.f,0.f,0.f,0.f,0.f};
  v8f oacc[8];
#pragma unroll
  for (int d = 0; d < 8; ++d) oacc[d] = vzero;
  float rmax[8], rsum[8];
#pragma unroll
  for (int j = 0; j < 8; ++j) { rmax[j] = -1e30f; rsum[j] = 0.f; }

  const float scale = 0.07216878f;   // 1/sqrt(192)
  const int nchunks = (qb + 16 + 31) >> 5;

  for (int kc = 0; kc < nchunks; ++kc) {
    int kbase = kc << 5;
    v8f st[2];
    // scores: two 16x16 tiles, each = 6 WMMAs over the 192 feature dim
#pragma unroll
    for (int kt = 0; kt < 2; ++kt) {
      int kcol = kbase + kt * 16 + lane16;
      v8f sc = vzero;
      size_t kro = (size_t)(b * SS + kcol) * (N_HEADS * QK_DIM) + hIdx * QK_DIM;
#pragma unroll
      for (int c = 0; c < 6; ++c) {
        Frag kb;
        kb.u[0] = *(const u32x4*)(kf + kro + c * 32 + half * 8);
        kb.u[1] = *(const u32x4*)(kf + kro + c * 32 + 16 + half * 8);
        sc = __builtin_amdgcn_wmma_f32_16x16x32_bf16(
            false, qa[c].v, false, kb.v, (short)0, sc, false, false);
      }
#pragma unroll
      for (int j = 0; j < 8; ++j) {
        int mrow = qb + j + 8 * half;
        st[kt][j] = (kcol <= mrow) ? sc[j] * scale : -1e30f;
      }
    }

    // online softmax update (row stats across 16-lane groups)
    float corr[8];
#pragma unroll
    for (int j = 0; j < 8; ++j) {
      float lm = fmaxf(st[0][j], st[1][j]);
#pragma unroll
      for (int o = 1; o < 16; o <<= 1) lm = fmaxf(lm, __shfl_xor(lm, o, 32));
      float mnew = fmaxf(rmax[j], lm);
      corr[j] = __expf(rmax[j] - mnew);
      float p0 = __expf(st[0][j] - mnew);
      float p1 = __expf(st[1][j] - mnew);
      float ps = p0 + p1;
#pragma unroll
      for (int o = 1; o < 16; o <<= 1) ps += __shfl_xor(ps, o, 32);
      rsum[j] = rsum[j] * corr[j] + ps;
      rmax[j] = mnew;
      int r = j + 8 * half;
      Pw[r * 32 + lane16]      = f2bf(p0);
      Pw[r * 32 + 16 + lane16] = f2bf(p1);
    }
#pragma unroll
    for (int d = 0; d < 8; ++d)
#pragma unroll
      for (int j = 0; j < 8; ++j) oacc[d][j] *= corr[j];

    // P (C-layout) -> A-layout via LDS round trip (wave-private region)
    asm volatile("s_wait_dscnt 0" ::: "memory");
    Frag pa;
    pa.u[0] = *(const u32x4*)(&Pw[lane16 * 32 + half * 8]);
    pa.u[1] = *(const u32x4*)(&Pw[lane16 * 32 + 16 + half * 8]);

    // P @ V : 8 WMMAs over head dim, V read k-contiguous from vt[d][.]
#pragma unroll
    for (int d = 0; d < 8; ++d) {
      size_t vro = (size_t)(hIdx * HEAD_DIM + d * 16 + lane16) * BS
                   + (size_t)b * SS + kbase;
      Frag vb;
      vb.u[0] = *(const u32x4*)(vt + vro + half * 8);
      vb.u[1] = *(const u32x4*)(vt + vro + 16 + half * 8);
      oacc[d] = __builtin_amdgcn_wmma_f32_16x16x32_bf16(
          false, pa.v, false, vb.v, (short)0, oacc[d], false, false);
    }
  }

  float inv[8];
#pragma unroll
  for (int j = 0; j < 8; ++j) inv[j] = 1.0f / rsum[j];
#pragma unroll
  for (int d = 0; d < 8; ++d) {
    int col = hIdx * HEAD_DIM + d * 16 + lane16;
#pragma unroll
    for (int j = 0; j < 8; ++j) {
      int m = qb + j + 8 * half;
      attn[(size_t)(b * SS + m) * D_MODEL + col] = f2bf(oacc[d][j] * inv[j]);
    }
  }
}

/* ------------------------------------------------------------------ */
extern "C" void kernel_launch(void* const* d_in, const int* in_sizes, int n_in,
                              void* d_out, int out_size, void* d_ws, size_t ws_size,
                              hipStream_t stream) {
  (void)in_sizes; (void)n_in; (void)out_size; (void)ws_size;
  const float* x     = (const float*)d_in[0];
  const float* w_dkv = (const float*)d_in[1];
  const float* b_dkv = (const float*)d_in[2];
  const float* w_dq  = (const float*)d_in[3];
  const float* b_dq  = (const float*)d_in[4];
  const float* w_rk  = (const float*)d_in[5];
  const float* b_rk  = (const float*)d_in[6];
  const float* w_rq  = (const float*)d_in[7];
  const float* b_rq  = (const float*)d_in[8];
  const float* w_uk  = (const float*)d_in[9];
  const float* b_uk  = (const float*)d_in[10];
  const float* w_uv  = (const float*)d_in[11];
  const float* b_uv  = (const float*)d_in[12];
  const float* w_uq  = (const float*)d_in[13];
  const float* b_uq  = (const float*)d_in[14];
  const float* w_o   = (const float*)d_in[15];
  const float* b_o   = (const float*)d_in[16];

  unsigned short* ws = (unsigned short*)d_ws;
  unsigned short* xb   = ws; ws += (size_t)BS * D_MODEL;          // x in bf16
  unsigned short* lkv  = ws; ws += (size_t)BS * KV_DIM;           // latent_kv
  unsigned short* lq   = ws; ws += (size_t)BS * Q_DIM;            // latent_q
  unsigned short* qfb  = ws; ws += (size_t)BS * N_HEADS * QK_DIM; // [s][h][192]
  unsigned short* kfb  = ws; ws += (size_t)BS * N_HEADS * QK_DIM; // [s][h][192]
  unsigned short* vtb  = ws; ws += (size_t)D_MODEL * BS;          // V^T [d][b*S+s]
  unsigned short* rkr  = ws; ws += (size_t)BS * ROPE_DIM;         // raw rope_k
  unsigned short* attn = ws; ws += (size_t)BS * D_MODEL;          // attention out

  cvt_f32_bf16<<<(BS * D_MODEL) / 256, 256, 0, stream>>>(x, xb, BS * D_MODEL);

  // latent_kv = x @ w_dkv + b_dkv
  gemm_bf16<<<dim3(KV_DIM / 128, BS / 128), 256, 0, stream>>>(
      xb, D_MODEL, w_dkv, KV_DIM, b_dkv, lkv, KV_DIM,
      BS, KV_DIM, D_MODEL, KV_DIM, KV_DIM, 0, 0);
  // latent_q = x @ w_dq + b_dq
  gemm_bf16<<<dim3(Q_DIM / 128, BS / 128), 256, 0, stream>>>(
      xb, D_MODEL, w_dq, Q_DIM, b_dq, lq, Q_DIM,
      BS, Q_DIM, D_MODEL, Q_DIM, Q_DIM, 0, 0);
  // rope_k raw = x @ w_rk + b_rk
  gemm_bf16<<<dim3(1, BS / 128), 256, 0, stream>>>(
      xb, D_MODEL, w_rk, ROPE_DIM, b_rk, rkr, ROPE_DIM,
      BS, ROPE_DIM, D_MODEL, ROPE_DIM, ROPE_DIM, 0, 0);
  // q -> qf[:, h, 0:128]
  gemm_bf16<<<dim3(D_MODEL / 128, BS / 128), 256, 0, stream>>>(
      lq, Q_DIM, w_uq, D_MODEL, b_uq, qfb, N_HEADS * QK_DIM,
      BS, D_MODEL, Q_DIM, HEAD_DIM, QK_DIM, 0, 0);
  // rope_q raw -> qf[:, h, 128:192]
  gemm_bf16<<<dim3((N_HEADS * ROPE_DIM) / 128, BS / 128), 256, 0, stream>>>(
      lq, Q_DIM, w_rq, N_HEADS * ROPE_DIM, b_rq, qfb, N_HEADS * QK_DIM,
      BS, N_HEADS * ROPE_DIM, Q_DIM, ROPE_DIM, QK_DIM, HEAD_DIM, 0);
  rope_q<<<(BS * N_HEADS * 32) / 256, 256, 0, stream>>>(qfb);
  // k -> kf[:, h, 0:128]
  gemm_bf16<<<dim3(D_MODEL / 128, BS / 128), 256, 0, stream>>>(
      lkv, KV_DIM, w_uk, D_MODEL, b_uk, kfb, N_HEADS * QK_DIM,
      BS, D_MODEL, KV_DIM, HEAD_DIM, QK_DIM, 0, 0);
  rope_k_bcast<<<(BS * 32) / 256, 256, 0, stream>>>(rkr, kfb);
  // V^T (transposed store)
  gemm_bf16<<<dim3(D_MODEL / 128, BS / 128), 256, 0, stream>>>(
      lkv, KV_DIM, w_uv, D_MODEL, b_uv, vtb, BS,
      BS, D_MODEL, KV_DIM, 1, 1, 0, 1);
  // fused causal attention
  mla_attn<<<dim3(SS / 128, N_HEADS, BB), 256, 0, stream>>>(qfb, kfb, vtb, attn);
  // out = attn @ w_o + b_o  (fp32)
  gemm_bf16<<<dim3(D_MODEL / 128, BS / 128), 256, 0, stream>>>(
      attn, D_MODEL, w_o, D_MODEL, b_o, d_out, D_MODEL,
      BS, D_MODEL, D_MODEL, D_MODEL, D_MODEL, 0, 2);
}
